// GraphClassifier_61478161875419
// MI455X (gfx1250) — matrix operand order repaired and checked by reading, not compiled, not run on hardware
//
#include <hip/hip_runtime.h>

// ---------- types for WMMA ----------
typedef __attribute__((ext_vector_type(16))) __bf16 v16bf;
typedef __attribute__((ext_vector_type(8)))  float  v8f;

union BF16x16 { v16bf v; uint4 q[2]; };

#define FD 128
#define HD 128
#define NC 10

__device__ __forceinline__ unsigned short f2bf(float f) {
    unsigned int u = __float_as_uint(f);
    u += 0x7fffu + ((u >> 16) & 1u);   // round-to-nearest-even
    return (unsigned short)(u >> 16);
}

// ---------- degree / dinv ----------
__global__ void init_deg_kernel(float* deg, int N) {
    int i = blockIdx.x * blockDim.x + threadIdx.x;
    if (i < N) deg[i] = 1.0f;                       // self-loop
}
__global__ void acc_deg_kernel(const int* __restrict__ ei, float* deg, int E) {
    int e = blockIdx.x * blockDim.x + threadIdx.x;
    if (e < E) atomicAdd(&deg[ei[E + e]], 1.0f);    // dst row
}
__global__ void fin_dinv_kernel(float* deg, int N) {
    int i = blockIdx.x * blockDim.x + threadIdx.x;
    if (i < N) deg[i] = rsqrtf(fmaxf(deg[i], 1.0f));
}

// ---------- fp32 -> bf16 conversion ----------
__global__ void cvt_bf16_kernel(const float* __restrict__ in, unsigned short* __restrict__ out, size_t n) {
    size_t i = (size_t)blockIdx.x * blockDim.x + threadIdx.x;
    if (i < n) out[i] = f2bf(in[i]);
}
// W[k][n] (row-major [128][128]) -> Wt[n][k] bf16
__global__ void cvt_w_t_kernel(const float* __restrict__ W, unsigned short* __restrict__ Wt) {
    int t = blockIdx.x * blockDim.x + threadIdx.x;   // 16384
    int n = t >> 7, k = t & 127;
    Wt[t] = f2bf(W[k * HD + n]);
}

// ---------- WMMA GEMM: Y[N x 128] = Xb[N x 128] @ W  (Wt is [n][k] bf16) ----------
// 64 rows per block, 8 waves; each wave owns a 16-col slab, caches its 4 B
// fragments in VGPRs, and sweeps 4 row-tiles (16 WMMAs/wave).
__global__ __launch_bounds__(256) void gemm_bf16_kernel(const unsigned short* __restrict__ Xb,
                                                        const unsigned short* __restrict__ Wt,
                                                        float* __restrict__ Y, int N) {
    __shared__ unsigned short sA[64 * 128];     // 16 KB
    __shared__ unsigned short sB[128 * 128];    // 32 KB, [n][k]
    const int t  = threadIdx.x;
    const int R0 = blockIdx.x * 64;

    // cooperative load of Wt into LDS (2048 uint4, 8 per thread)
    {
        const uint4* src = (const uint4*)Wt;
        uint4* dst = (uint4*)sB;
#pragma unroll
        for (int i = 0; i < 8; ++i) dst[t + i * 256] = src[t + i * 256];
    }
    // cooperative load of 64-row A slab (1024 uint4, 4 per thread)
    {
#pragma unroll
        for (int i = 0; i < 4; ++i) {
            int q  = t + i * 256;           // uint4 index
            int r  = q >> 4;                // row 0..63
            int c8 = (q & 15) * 8;          // elem col
            uint4 v = make_uint4(0u, 0u, 0u, 0u);
            if (R0 + r < N) v = *(const uint4*)(Xb + (size_t)(R0 + r) * 128 + c8);
            *(uint4*)(sA + r * 128 + c8) = v;
        }
    }
    __syncthreads();

    const int w    = t >> 5;        // wave 0..7 -> 16-col slab
    const int lane = t & 31;
    const int m    = lane & 15;     // row-in-tile (A) / col-in-tile (B,C)
    const int hi   = lane >> 4;     // half-wave K split

    // cache the 4 B fragments (K = 0..127) in registers
    BF16x16 bf[4];
    {
        const unsigned short* pb0 = sB + (w * 16 + m) * 128 + hi * 16;
#pragma unroll
        for (int kc = 0; kc < 4; ++kc) {
            const unsigned short* pb = pb0 + kc * 32;
            bf[kc].q[0] = *(const uint4*)(pb);
            bf[kc].q[1] = *(const uint4*)(pb + 8);
        }
    }

    const int  col      = w * 16 + m;
    const bool fulltile = (R0 + 64 <= N);

#pragma unroll
    for (int rt = 0; rt < 4; ++rt) {
        v8f c = {};
        const unsigned short* pa0 = sA + (rt * 16 + m) * 128 + hi * 16;
#pragma unroll
        for (int kc = 0; kc < 4; ++kc) {
            // A 16x32 bf16: VGPR0-7 = K[koff..koff+15] (koff = hi*16), matching B
            BF16x16 a;
            const unsigned short* pa = pa0 + kc * 32;
            a.q[0] = *(const uint4*)(pa);
            a.q[1] = *(const uint4*)(pa + 8);
            c = __builtin_amdgcn_wmma_f32_16x16x32_bf16(false, a.v, false, bf[kc].v,
                                                        (short)0, c, false, false);
        }
        // C layout: lane 0-15 -> M=vi, N=lane ; lane 16-31 -> M=vi+8, N=lane-16
        float* yp = Y + (size_t)(R0 + rt * 16 + hi * 8) * 128 + col;
        if (fulltile) {
#pragma unroll
            for (int vi = 0; vi < 8; ++vi) yp[(size_t)vi * 128] = c[vi];
        } else {
            int rbase = R0 + rt * 16 + hi * 8;
#pragma unroll
            for (int vi = 0; vi < 8; ++vi)
                if (rbase + vi < N) yp[(size_t)vi * 128] = c[vi];
        }
    }
}

// ---------- H = b + M * dinv^2  (bias + self-loop) ----------
__global__ void init_h_kernel(const float* __restrict__ M, const float* __restrict__ dinv,
                              const float* __restrict__ b, float* __restrict__ H, size_t total) {
    size_t t = (size_t)blockIdx.x * blockDim.x + threadIdx.x;
    if (t < total) {
        int f = (int)(t & 127);
        size_t i = t >> 7;
        float di = dinv[i];
        H[t] = b[f] + M[t] * di * di;
    }
}

// ---------- edge scatter-add: one wave32 per edge, float4 per lane ----------
__global__ __launch_bounds__(256) void aggregate_kernel(const float* __restrict__ M,
                                                        const int* __restrict__ ei,
                                                        const float* __restrict__ dinv,
                                                        float* __restrict__ H, int E) {
    int e = blockIdx.x * 8 + (threadIdx.x >> 5);
    if (e >= E) return;
    int lane = threadIdx.x & 31;
    int s = ei[e], d = ei[E + e];
    float nm = dinv[s] * dinv[d];
    float4 v = *(const float4*)(M + (size_t)s * 128 + lane * 4);
    float* hp = H + (size_t)d * 128 + lane * 4;
    atomicAdd(hp + 0, v.x * nm);
    atomicAdd(hp + 1, v.y * nm);
    atomicAdd(hp + 2, v.z * nm);
    atomicAdd(hp + 3, v.w * nm);
}

__global__ void relu_kernel(float* H, size_t n) {
    size_t i = (size_t)blockIdx.x * blockDim.x + threadIdx.x;
    if (i < n) H[i] = fmaxf(H[i], 0.0f);
}

// ---------- pooling ----------
__global__ void pool_init_kernel(float* mean, int* maxv, float* cnt, int G) {
    int t = blockIdx.x * blockDim.x + threadIdx.x;
    if (t < G * 128) { mean[t] = 0.0f; maxv[t] = 0; }   // 0 bits == 0.0f
    if (t < G) cnt[t] = 0.0f;
}
__global__ __launch_bounds__(256) void pool_node_kernel(const float* __restrict__ H,
                                                        const int* __restrict__ batch,
                                                        float* mean, int* maxv, int N) {
    int i = blockIdx.x * 8 + (threadIdx.x >> 5);
    if (i >= N) return;
    int lane = threadIdx.x & 31;
    int g = batch[i];
    float4 v = *(const float4*)(H + (size_t)i * 128 + lane * 4);
    float* mp = mean + (size_t)g * 128 + lane * 4;
    int*   xp = maxv + (size_t)g * 128 + lane * 4;
    atomicAdd(mp + 0, v.x); atomicAdd(mp + 1, v.y);
    atomicAdd(mp + 2, v.z); atomicAdd(mp + 3, v.w);
    // post-ReLU values are >= 0, so int-bit compare preserves float order
    atomicMax(xp + 0, __float_as_int(v.x));
    atomicMax(xp + 1, __float_as_int(v.y));
    atomicMax(xp + 2, __float_as_int(v.z));
    atomicMax(xp + 3, __float_as_int(v.w));
}
__global__ void pool_cnt_kernel(const int* __restrict__ batch, float* cnt, int N) {
    int i = blockIdx.x * blockDim.x + threadIdx.x;
    if (i < N) atomicAdd(&cnt[batch[i]], 1.0f);
}
__global__ void pool_fin_kernel(float* mean, const float* __restrict__ cnt, int G) {
    int t = blockIdx.x * blockDim.x + threadIdx.x;
    if (t < G * 128) mean[t] /= fmaxf(cnt[t >> 7], 1.0f);
}

// ---------- FC head + log_softmax : one block per graph ----------
__global__ __launch_bounds__(128) void fc_kernel(const float* __restrict__ mean,
                                                 const int* __restrict__ maxv,
                                                 const float* __restrict__ fcW1, const float* __restrict__ fcb1,
                                                 const float* __restrict__ fcW2, const float* __restrict__ fcb2,
                                                 float* __restrict__ out) {
    int g = blockIdx.x, t = threadIdx.x;
    __shared__ float z[256];
    __shared__ float h1[128];
    __shared__ float logit[NC];
    __shared__ float red[2];
    z[t]       = mean[g * 128 + t];
    z[t + 128] = __int_as_float(maxv[g * 128 + t]);
    __syncthreads();
    float acc = fcb1[t];
#pragma unroll 8
    for (int k = 0; k < 256; ++k) acc += z[k] * fcW1[k * HD + t];
    h1[t] = fmaxf(acc, 0.0f);
    __syncthreads();
    if (t < NC) {
        float a = fcb2[t];
#pragma unroll 8
        for (int k = 0; k < 128; ++k) a += h1[k] * fcW2[k * NC + t];
        logit[t] = a;
    }
    __syncthreads();
    if (t == 0) {
        float mx = logit[0];
        for (int i = 1; i < NC; ++i) mx = fmaxf(mx, logit[i]);
        float s = 0.0f;
        for (int i = 0; i < NC; ++i) s += expf(logit[i] - mx);
        red[0] = mx; red[1] = logf(s);
    }
    __syncthreads();
    if (t < NC) out[g * NC + t] = logit[t] - red[0] - red[1];
}

// ======================================================================
extern "C" void kernel_launch(void* const* d_in, const int* in_sizes, int n_in,
                              void* d_out, int out_size, void* d_ws, size_t ws_size,
                              hipStream_t stream) {
    const float* x     = (const float*)d_in[0];
    const int*   ei    = (const int*)  d_in[1];    // [2][E]
    const int*   batch = (const int*)  d_in[2];
    const float* W1    = (const float*)d_in[4];
    const float* b1    = (const float*)d_in[5];
    const float* W2    = (const float*)d_in[6];
    const float* b2    = (const float*)d_in[7];
    const float* fcW1  = (const float*)d_in[8];
    const float* fcb1  = (const float*)d_in[9];
    const float* fcW2  = (const float*)d_in[10];
    const float* fcb2  = (const float*)d_in[11];
    float* out = (float*)d_out;

    const int N = in_sizes[0] / FD;
    const int E = in_sizes[1] / 2;
    const int G = out_size / NC;
    const size_t NF = (size_t)N * 128;

    // ---- workspace carve-up (256B aligned) ----
    char* base = (char*)d_ws;
    size_t off = 0;
    auto carve = [&](size_t bytes) -> char* {
        char* p = base + off;
        off = (off + bytes + 255) & ~(size_t)255;
        return p;
    };
    float*          dinv = (float*)         carve((size_t)N * 4);
    unsigned short* Xb   = (unsigned short*)carve(NF * 2);
    float*          Mb   = (float*)         carve(NF * 4);
    float*          Hb   = (float*)         carve(NF * 4);
    unsigned short* Wt1  = (unsigned short*)carve(128 * 128 * 2);
    unsigned short* Wt2  = (unsigned short*)carve(128 * 128 * 2);
    float*          mean = (float*)         carve((size_t)G * 128 * 4);
    int*            maxv = (int*)           carve((size_t)G * 128 * 4);
    float*          cnt  = (float*)         carve((size_t)G * 4);
    (void)ws_size;

    const int T = 256;
    auto cdiv = [](size_t a, size_t b) -> unsigned { return (unsigned)((a + b - 1) / b); };

    // degrees -> dinv
    init_deg_kernel<<<cdiv(N, T), T, 0, stream>>>(dinv, N);
    acc_deg_kernel<<<cdiv(E, T), T, 0, stream>>>(ei, dinv, E);
    fin_dinv_kernel<<<cdiv(N, T), T, 0, stream>>>(dinv, N);

    // transpose+convert weights once
    cvt_w_t_kernel<<<cdiv(128 * 128, T), T, 0, stream>>>(W1, Wt1);
    cvt_w_t_kernel<<<cdiv(128 * 128, T), T, 0, stream>>>(W2, Wt2);

    const unsigned gemm_blocks = cdiv(N, 64);
    const unsigned agg_blocks  = cdiv(E, 8);

    // ---- layer 1 ----
    cvt_bf16_kernel<<<cdiv(NF, T), T, 0, stream>>>(x, Xb, NF);
    gemm_bf16_kernel<<<gemm_blocks, T, 0, stream>>>(Xb, Wt1, Mb, N);
    init_h_kernel<<<cdiv(NF, T), T, 0, stream>>>(Mb, dinv, b1, Hb, NF);
    aggregate_kernel<<<agg_blocks, T, 0, stream>>>(Mb, ei, dinv, Hb, E);
    relu_kernel<<<cdiv(NF, T), T, 0, stream>>>(Hb, NF);

    // ---- layer 2 ----
    cvt_bf16_kernel<<<cdiv(NF, T), T, 0, stream>>>(Hb, Xb, NF);
    gemm_bf16_kernel<<<gemm_blocks, T, 0, stream>>>(Xb, Wt2, Mb, N);
    init_h_kernel<<<cdiv(NF, T), T, 0, stream>>>(Mb, dinv, b2, Hb, NF);
    aggregate_kernel<<<agg_blocks, T, 0, stream>>>(Mb, ei, dinv, Hb, E);
    relu_kernel<<<cdiv(NF, T), T, 0, stream>>>(Hb, NF);

    // ---- pooling ----
    pool_init_kernel<<<cdiv((size_t)G * 128, T), T, 0, stream>>>(mean, maxv, cnt, G);
    pool_node_kernel<<<cdiv(N, 8), T, 0, stream>>>(Hb, batch, mean, maxv, N);
    pool_cnt_kernel<<<cdiv(N, T), T, 0, stream>>>(batch, cnt, N);
    pool_fin_kernel<<<cdiv((size_t)G * 128, T), T, 0, stream>>>(mean, cnt, G);

    // ---- FC head ----
    fc_kernel<<<G, 128, 0, stream>>>(mean, maxv, fcW1, fcb1, fcW2, fcb2, out);
}
